// Generator_72894184948121
// MI455X (gfx1250) — matrix operand order repaired and checked by reading, not compiled, not run on hardware
//
#include <hip/hip_runtime.h>
#include <math.h>

#define BB 32
#define BEAM 4
#define VV 32000
#define HH 1024
#define DD 512
#define LL 20
#define NBEAMS (BB * BEAM)   // 128
#define G4H (4 * HH)         // 4096

typedef __attribute__((ext_vector_type(16))) __bf16 v16bf;
typedef __attribute__((ext_vector_type(8)))  float  v8f;

union BF16Frag { v16bf v; unsigned short s[16]; uint4 q[2]; };

__device__ __forceinline__ unsigned short f2bf(float f) {
  unsigned int u = __float_as_uint(f);
  u += 0x7fffu + ((u >> 16) & 1u);      // round-to-nearest-even
  return (unsigned short)(u >> 16);
}

// Packed fragment layouts (one 512-element bf16 block per 16x32 A-tile / 32x16 B-tile;
// lane-major so a lane's 16 elements are 32 contiguous bytes -> two b128 loads).
__device__ __forceinline__ size_t packA_idx(int row, int k, int K) {
  int mt = row >> 4, ml = row & 15;
  int kt = k >> 5, ko = k & 31;
  int half = ko >> 4;
  int e = ko & 15;
  int lane = (half << 4) + ml;
  return (((size_t)mt * (K >> 5) + kt) * 32 + lane) * 16 + e;
}
__device__ __forceinline__ size_t packB_idx(int k, int n, int K) {
  int kt = k >> 5, ko = k & 31;
  int nt = n >> 4, nn = n & 15;
  int half = ko >> 4, e = ko & 15;
  return (((size_t)nt * (K >> 5) + kt) * 32 + ((half << 4) + nn)) * 16 + e;
}

// ---------------- weight pack: f32 [K x N] row-major -> packed bf16 B ----------------
__global__ void pack_b_kernel(const float* __restrict__ src,
                              unsigned short* __restrict__ dst, int K, int N) {
  long total = (long)K * N;
  long i = (long)blockIdx.x * blockDim.x + threadIdx.x;
  long stride = (long)gridDim.x * blockDim.x;
  for (; i < total; i += stride) {
    int k = (int)(i / N), n = (int)(i - (long)k * N);
    dst[packB_idx(k, n, K)] = f2bf(src[i]);
  }
}

// ---------------- activation pack: f32 [rows x K] row-major -> packed bf16 A ----------------
__global__ void pack_a_f32_kernel(const float* __restrict__ src,
                                  unsigned short* __restrict__ dst, int K, long total) {
  long i = (long)blockIdx.x * blockDim.x + threadIdx.x;
  long stride = (long)gridDim.x * blockDim.x;
  for (; i < total; i += stride) {
    int row = (int)(i / K), k = (int)(i - (long)row * K);
    dst[packA_idx(row, k, K)] = f2bf(src[i]);
  }
}

// ---------------- init beam scalar state ----------------
__global__ void init_state_kernel(int* stops, float* prob, int* wl) {
  int i = threadIdx.x;
  if (i < NBEAMS) { stops[i] = 0; prob[i] = 1.0f; wl[i] = 0; }
}

// ---------------- embedding gather (f32 -> packed bf16 A rows) ----------------
__global__ void gather_embed_kernel(const int* __restrict__ words,
                                    const float* __restrict__ E,
                                    unsigned short* __restrict__ xb) {
  int row = blockIdx.x;
  int w = words[row];
  const float* e = E + (long)w * DD;
  for (int i = threadIdx.x; i < DD; i += blockDim.x)
    xb[packA_idx(row, i, DD)] = f2bf(e[i]);
}

// ---------------- WMMA bf16 GEMM on packed fragments ----------------
// C[M x N] f32 = A0*B0 (+ A1*B1) (+bias). block = 256 (8 wave32s);
// each wave computes a 16x32 tile (2 accumulators, A reused).
__global__ void gemm_bf16_kernel(const unsigned short* __restrict__ A0,
                                 const unsigned short* __restrict__ B0, int K0,
                                 const unsigned short* __restrict__ A1,
                                 const unsigned short* __restrict__ B1, int K1,
                                 const float* __restrict__ bias,
                                 float* __restrict__ C, int N) {
  const int lane = threadIdx.x & 31;
  const int wave = threadIdx.x >> 5;
  const int mt   = blockIdx.x;
  const int nt0  = ((blockIdx.y << 3) + wave) << 1;   // two 16-col tiles per wave
  const int l15  = lane & 15;
  const int half = lane >> 4;

  v8f acc0 = {}, acc1 = {};

  {
    const int KT = K0 >> 5;
    const uint4* a  = (const uint4*)(A0 + ((size_t)mt * KT * 32 + lane) * 16);
    const uint4* b0 = (const uint4*)(B0 + ((size_t)nt0 * KT * 32 + lane) * 16);
    const uint4* b1 = (const uint4*)(B0 + ((size_t)(nt0 + 1) * KT * 32 + lane) * 16);
    for (int kt = 0; kt < KT; ++kt) {
      BF16Frag af, bf0, bf1;
      af.q[0]  = a[0];  af.q[1]  = a[1];
      bf0.q[0] = b0[0]; bf0.q[1] = b0[1];
      bf1.q[0] = b1[0]; bf1.q[1] = b1[1];
      acc0 = __builtin_amdgcn_wmma_f32_16x16x32_bf16(false, af.v, false, bf0.v,
                                                     (short)0, acc0, false, false);
      acc1 = __builtin_amdgcn_wmma_f32_16x16x32_bf16(false, af.v, false, bf1.v,
                                                     (short)0, acc1, false, false);
      a += 64; b0 += 64; b1 += 64;   // 512 bf16 per block = 64 uint4
    }
  }
  if (A1) {
    const int KT = K1 >> 5;
    const uint4* a  = (const uint4*)(A1 + ((size_t)mt * KT * 32 + lane) * 16);
    const uint4* b0 = (const uint4*)(B1 + ((size_t)nt0 * KT * 32 + lane) * 16);
    const uint4* b1 = (const uint4*)(B1 + ((size_t)(nt0 + 1) * KT * 32 + lane) * 16);
    for (int kt = 0; kt < KT; ++kt) {
      BF16Frag af, bf0, bf1;
      af.q[0]  = a[0];  af.q[1]  = a[1];
      bf0.q[0] = b0[0]; bf0.q[1] = b0[1];
      bf1.q[0] = b1[0]; bf1.q[1] = b1[1];
      acc0 = __builtin_amdgcn_wmma_f32_16x16x32_bf16(false, af.v, false, bf0.v,
                                                     (short)0, acc0, false, false);
      acc1 = __builtin_amdgcn_wmma_f32_16x16x32_bf16(false, af.v, false, bf1.v,
                                                     (short)0, acc1, false, false);
      a += 64; b0 += 64; b1 += 64;
    }
  }

  const int m_base = (mt << 4) + (half << 3);
#pragma unroll
  for (int r = 0; r < 8; ++r) {
    int m = m_base + r;
    int n0 = (nt0 << 4) + l15;
    int n1 = n0 + 16;
    float v0 = acc0[r], v1 = acc1[r];
    if (bias) { v0 += bias[n0]; v1 += bias[n1]; }
    C[(long)m * N + n0] = v0;
    C[(long)m * N + n1] = v1;
  }
}

// ---------------- LSTM cell elementwise (torch gate order i,f,g,o) ----------------
__global__ void lstm_cell_kernel(const float* __restrict__ gates,
                                 const float* __restrict__ bih,
                                 const float* __restrict__ bhh,
                                 const float* __restrict__ cprev,
                                 float* __restrict__ h2, float* __restrict__ c2,
                                 unsigned short* __restrict__ h2b, int rows) {
  int idx = blockIdx.x * blockDim.x + threadIdx.x;
  if (idx >= rows * HH) return;
  int m = idx >> 10, j = idx & (HH - 1);
  const float* g = gates + (long)m * G4H;
  float gi = g[j]          + bih[j]          + bhh[j];
  float gf = g[j + HH]     + bih[j + HH]     + bhh[j + HH];
  float gg = g[j + 2 * HH] + bih[j + 2 * HH] + bhh[j + 2 * HH];
  float go = g[j + 3 * HH] + bih[j + 3 * HH] + bhh[j + 3 * HH];
  float si = 1.0f / (1.0f + expf(-gi));
  float sf = 1.0f / (1.0f + expf(-gf));
  float so = 1.0f / (1.0f + expf(-go));
  float cv = sf * cprev[idx] + si * tanhf(gg);
  float hv = so * tanhf(cv);
  c2[idx] = cv;
  h2[idx] = hv;
  h2b[packA_idx(m, j, HH)] = f2bf(hv);
}

// ---------------- per-row max + log-sum-exp over V ----------------
__global__ void row_stats_kernel(const float* __restrict__ logits,
                                 float* __restrict__ rowmax,
                                 float* __restrict__ rowlse) {
  int r = blockIdx.x;
  const float* row = logits + (long)r * VV;
  __shared__ float red[256];
  float mx = -3.4e38f;
  for (int v = threadIdx.x; v < VV; v += 256) mx = fmaxf(mx, row[v]);
  red[threadIdx.x] = mx; __syncthreads();
  for (int s = 128; s > 0; s >>= 1) {
    if (threadIdx.x < s) red[threadIdx.x] = fmaxf(red[threadIdx.x], red[threadIdx.x + s]);
    __syncthreads();
  }
  mx = red[0]; __syncthreads();
  float sum = 0.0f;
  for (int v = threadIdx.x; v < VV; v += 256) sum += expf(row[v] - mx);
  red[threadIdx.x] = sum; __syncthreads();
  for (int s = 128; s > 0; s >>= 1) {
    if (threadIdx.x < s) red[threadIdx.x] += red[threadIdx.x + s];
    __syncthreads();
  }
  if (threadIdx.x == 0) { rowmax[r] = mx; rowlse[r] = logf(red[0]); }
}

// ---------------- score + per-batch top-4 + scalar-state update ----------------
__global__ void topk_update_kernel(const float* __restrict__ logits,
                                   const float* __restrict__ rowmax,
                                   const float* __restrict__ rowlse,
                                   const int* __restrict__ st_in,
                                   const float* __restrict__ pr_in,
                                   const int* __restrict__ wl_in,
                                   int* __restrict__ st_out,
                                   float* __restrict__ pr_out,
                                   int* __restrict__ wl_out,
                                   int* __restrict__ words_cur,
                                   int* __restrict__ words_hist_t,
                                   int* __restrict__ prevs_hist_t,
                                   int beam_in) {
  const float C6 = 2.93015605f;          // 6^0.6
  int b = blockIdx.x;
  int tid = threadIdx.x;
  int base = b * beam_in;
  int total = beam_in * VV;

  float bs[4]; int bi[4];
#pragma unroll
  for (int k = 0; k < 4; ++k) { bs[k] = -3.4e38f; bi[k] = 0x7fffffff; }

  for (int c = tid; c < total; c += 256) {
    int r = base + c / VV;
    int v = c - (c / VV) * VV;
    int st = st_in[r];
    float wp;
    if (st) wp = (v == 0) ? pr_in[r] : 0.0f;
    else    wp = expf(logits[(long)r * VV + v] - rowmax[r] - rowlse[r]) * pr_in[r];
    float lw = logf(fminf(fmaxf(wp, 1e-20f), 1.0f));
    int ext = ((v != 0) && (v != 2)) ? (1 - st) : 0;
    int wlm = wl_in[r] + ext;
    float lp = powf((float)(wlm + 5), 0.6f) / C6;
    float s = lw / lp;
    int cc = c;
#pragma unroll
    for (int k = 0; k < 4; ++k) {
      bool better = (s > bs[k]) || (s == bs[k] && cc < bi[k]);
      if (better) { float ts = bs[k]; int ti = bi[k]; bs[k] = s; bi[k] = cc; s = ts; cc = ti; }
    }
  }

  __shared__ float ssc[1024];
  __shared__ int   sid[1024];
#pragma unroll
  for (int k = 0; k < 4; ++k) { ssc[tid * 4 + k] = bs[k]; sid[tid * 4 + k] = bi[k]; }
  __syncthreads();

  if (tid == 0) {
    float fs[4]; int fi[4];
#pragma unroll
    for (int k = 0; k < 4; ++k) { fs[k] = -3.4e38f; fi[k] = 0x7fffffff; }
    for (int e = 0; e < 1024; ++e) {
      float s = ssc[e]; int cc = sid[e];
#pragma unroll
      for (int k = 0; k < 4; ++k) {
        bool better = (s > fs[k]) || (s == fs[k] && cc < fi[k]);
        if (better) { float ts = fs[k]; int ti = fi[k]; fs[k] = s; fi[k] = cc; s = ts; cc = ti; }
      }
    }
    for (int j = 0; j < 4; ++j) {
      int c = fi[j];
      int prev = base + c / VV;
      int nw = c - (c / VV) * VV;
      int row = b * BEAM + j;
      int stp = st_in[prev];
      float p2;
      if (stp) p2 = (nw == 0) ? pr_in[prev] : 0.0f;
      else     p2 = expf(logits[(long)prev * VV + nw] - rowmax[prev] - rowlse[prev]) * pr_in[prev];
      int st2 = (nw <= 2) ? 1 : 0;
      st_out[row] = st2;
      pr_out[row] = p2;
      wl_out[row] = wl_in[prev] + (1 - st2);
      words_cur[row]    = nw;
      words_hist_t[row] = nw;
      prevs_hist_t[row] = prev;
    }
  }
}

// ---------------- reorder h/c by prev, record outs, packed bf16 h for next step ----------------
__global__ void reorder_kernel(const float* __restrict__ h2,
                               const float* __restrict__ c2,
                               const int* __restrict__ prevs_t,
                               unsigned short* __restrict__ hs_b,
                               float* __restrict__ cstate,
                               float* __restrict__ outs_t) {
  int row = blockIdx.x;
  int prev = prevs_t[row];
  const float* hp = h2 + (long)prev * HH;
  const float* cp = c2 + (long)prev * HH;
  for (int i = threadIdx.x; i < HH; i += blockDim.x) {
    float hv = hp[i];
    hs_b[packA_idx(row, i, HH)] = f2bf(hv);
    cstate[row * HH + i]        = cp[i];
    outs_t[(long)row * HH + i]  = hv;
  }
}

// ---------------- backtrack beam 0 path ----------------
__global__ void backtrack_kernel(const int* __restrict__ words_hist,
                                 const int* __restrict__ prevs_hist,
                                 const float* __restrict__ outs_hist,
                                 float* __restrict__ out_words,
                                 float* __restrict__ out_h) {
  int b = blockIdx.x;
  __shared__ int chain[LL];
  if (threadIdx.x == 0) {
    int idx = b * BEAM;
    for (int t = LL - 1; t >= 0; --t) { chain[t] = idx; idx = prevs_hist[t * NBEAMS + idx]; }
  }
  __syncthreads();
  for (int t = 0; t < LL; ++t) {
    int idx = chain[t];
    if (threadIdx.x == 0)
      out_words[b * LL + t] = (float)words_hist[t * NBEAMS + idx];
    const float* src = outs_hist + ((long)t * NBEAMS + idx) * HH;
    float* dst = out_h + ((long)b * LL + t) * HH;
    for (int i = threadIdx.x; i < HH; i += blockDim.x) dst[i] = src[i];
  }
}

extern "C" void kernel_launch(void* const* d_in, const int* in_sizes, int n_in,
                              void* d_out, int out_size, void* d_ws, size_t ws_size,
                              hipStream_t stream) {
  (void)in_sizes; (void)n_in; (void)out_size; (void)ws_size;
  const int*   word = (const int*)  d_in[0];
  const float* h0   = (const float*)d_in[1];
  const float* c0   = (const float*)d_in[2];
  const float* E    = (const float*)d_in[3];
  const float* Wih  = (const float*)d_in[4];
  const float* Whh  = (const float*)d_in[5];
  const float* bih  = (const float*)d_in[6];
  const float* bhh  = (const float*)d_in[7];
  const float* Wout = (const float*)d_in[8];
  const float* bout = (const float*)d_in[9];
  // d_in[10] (length) is a device scalar; reference constant L=20 is baked in.

  char* p = (char*)d_ws;
  auto alloc = [&](size_t bytes) -> void* {
    void* r = (void*)p; p += (bytes + 255) & ~(size_t)255; return r;
  };

  unsigned short* Wih_b  = (unsigned short*)alloc((size_t)DD * G4H * 2);
  unsigned short* Whh_b  = (unsigned short*)alloc((size_t)HH * G4H * 2);
  unsigned short* Wout_b = (unsigned short*)alloc((size_t)HH * VV * 2);
  unsigned short* x_b    = (unsigned short*)alloc((size_t)NBEAMS * DD * 2);
  unsigned short* hs_b   = (unsigned short*)alloc((size_t)NBEAMS * HH * 2);
  unsigned short* h2_b   = (unsigned short*)alloc((size_t)NBEAMS * HH * 2);
  float* gates   = (float*)alloc((size_t)NBEAMS * G4H * 4);
  float* h2      = (float*)alloc((size_t)NBEAMS * HH * 4);
  float* c2      = (float*)alloc((size_t)NBEAMS * HH * 4);
  float* cstate  = (float*)alloc((size_t)NBEAMS * HH * 4);
  float* logits  = (float*)alloc((size_t)NBEAMS * VV * 4);
  float* rowmax  = (float*)alloc(NBEAMS * 4);
  float* rowlse  = (float*)alloc(NBEAMS * 4);
  int*   stopsA  = (int*)  alloc(NBEAMS * 4);
  int*   stopsB  = (int*)  alloc(NBEAMS * 4);
  float* probA   = (float*)alloc(NBEAMS * 4);
  float* probB   = (float*)alloc(NBEAMS * 4);
  int*   wlA     = (int*)  alloc(NBEAMS * 4);
  int*   wlB     = (int*)  alloc(NBEAMS * 4);
  int*   words_cur  = (int*)alloc(NBEAMS * 4);
  int*   words_hist = (int*)alloc((size_t)LL * NBEAMS * 4);
  int*   prevs_hist = (int*)alloc((size_t)LL * NBEAMS * 4);
  float* outs_hist  = (float*)alloc((size_t)LL * NBEAMS * HH * 4);

  // Per-launch bf16 weight conversion + fragment packing (deterministic, graph-safe).
  pack_b_kernel<<<4096, 256, 0, stream>>>(Wih,  Wih_b,  DD, G4H);
  pack_b_kernel<<<4096, 256, 0, stream>>>(Whh,  Whh_b,  HH, G4H);
  pack_b_kernel<<<4096, 256, 0, stream>>>(Wout, Wout_b, HH, VV);
  pack_a_f32_kernel<<<128, 256, 0, stream>>>(h0, hs_b, HH, (long)BB * HH);
  init_state_kernel<<<1, 128, 0, stream>>>(stopsA, probA, wlA);

  for (int t = 0; t < LL; ++t) {
    int N_in    = (t == 0) ? BB : NBEAMS;
    int beam_in = (t == 0) ? 1  : BEAM;
    int*   st_in  = (t & 1) ? stopsB : stopsA;
    int*   st_out = (t & 1) ? stopsA : stopsB;
    float* pr_in  = (t & 1) ? probB  : probA;
    float* pr_out = (t & 1) ? probA  : probB;
    int*   wl_in  = (t & 1) ? wlB    : wlA;
    int*   wl_out = (t & 1) ? wlA    : wlB;
    const int*   wsrc  = (t == 0) ? word : words_cur;
    const float* cprev = (t == 0) ? c0   : cstate;

    gather_embed_kernel<<<N_in, 256, 0, stream>>>(wsrc, E, x_b);

    // gates = x@Wih + h@Whh  : [N_in x 4096]
    gemm_bf16_kernel<<<dim3(N_in / 16, G4H / 256), 256, 0, stream>>>(
        x_b, Wih_b, DD, hs_b, Whh_b, HH, nullptr, gates, G4H);

    lstm_cell_kernel<<<(N_in * HH + 255) / 256, 256, 0, stream>>>(
        gates, bih, bhh, cprev, h2, c2, h2_b, N_in);

    // logits = h2@Wout + bout : [N_in x 32000]
    gemm_bf16_kernel<<<dim3(N_in / 16, VV / 256), 256, 0, stream>>>(
        h2_b, Wout_b, HH, nullptr, nullptr, 0, bout, logits, VV);

    row_stats_kernel<<<N_in, 256, 0, stream>>>(logits, rowmax, rowlse);

    topk_update_kernel<<<BB, 256, 0, stream>>>(
        logits, rowmax, rowlse, st_in, pr_in, wl_in, st_out, pr_out, wl_out,
        words_cur, words_hist + t * NBEAMS, prevs_hist + t * NBEAMS, beam_in);

    reorder_kernel<<<NBEAMS, 256, 0, stream>>>(
        h2, c2, prevs_hist + t * NBEAMS, hs_b, cstate,
        outs_hist + (size_t)t * NBEAMS * HH);
  }

  backtrack_kernel<<<BB, 256, 0, stream>>>(
      words_hist, prevs_hist, outs_hist,
      (float*)d_out, (float*)d_out + BB * LL);
}
